// RNNLayer_53515292508178
// MI455X (gfx1250) — matrix operand order repaired and checked by reading, not compiled, not run on hardware
//
#include <hip/hip_runtime.h>

// ---------------------------------------------------------------------------
// RNN layer for MI455X (gfx1250, wave32, WMMA)
//   B=64, T=2048, D=256, H=256, C=64
//   Phase 0: convert/transpose weights to bf16 [N][K] fragment-friendly layout
//   Phase 1: xw[T][B][H] = X @ W1 + b   (bf16 WMMA, f32 accum, 2 acc chains
//            per wave sharing one A fragment -> fills WMMA hazard slots)
//   Phase 2: sequential scan, 4 independent workgroups (16 batch rows each),
//            W2/V/h resident in LDS, dual-chain bf16 WMMA per step, fast tanh,
//            software-pipelined xw loads, global_prefetch of t+2 slab,
//            K-split logits over all 8 waves, 2 barriers per step.
// ---------------------------------------------------------------------------

#define B_DIM 64
#define T_DIM 2048
#define D_DIM 256
#define H_DIM 256
#define C_DIM 64

typedef __attribute__((ext_vector_type(16))) __bf16 v16bf;
typedef __attribute__((ext_vector_type(8)))  __bf16 v8bf;
typedef __attribute__((ext_vector_type(8)))  float  v8f;

#define WMMA_BF16(a, b, c) \
  __builtin_amdgcn_wmma_f32_16x16x32_bf16(false, (a), false, (b), (short)0, (c), false, false)

// Fast tanh on the recurrence critical path: 1 - 2/(e^{2x}+1).
// Saturates correctly: x>>0 -> exp=inf -> 1; x<<0 -> exp=0 -> -1.
__device__ __forceinline__ float fast_tanhf(float x) {
  const float e = __expf(2.0f * x);
  return 1.0f - 2.0f / (e + 1.0f);
}

// ---- Fragment loaders (layouts per CDNA5 ISA §7.12.2, wave32) ----

// A (16x32, 16-bit): lanes 0-15 -> M=lane, K = k0+{0..7, 16..23}
//                    lanes 16-31 -> M=lane-16, K = k0+{8..15, 24..31}
__device__ __forceinline__ v16bf load_a_f32(const float* __restrict__ rowPtr,
                                            int k0, int lane) {
  const int kb = (lane & 16) ? 8 : 0;
  const float* p = rowPtr + k0 + kb;
  v16bf a;
#pragma unroll
  for (int e = 0; e < 8; ++e) a[e] = (__bf16)p[e];
#pragma unroll
  for (int e = 0; e < 8; ++e) a[8 + e] = (__bf16)p[16 + e];
  return a;
}

__device__ __forceinline__ v16bf load_a_bf16(const __bf16* rowPtr, int k0, int lane) {
  const int kb = (lane & 16) ? 8 : 0;
  v8bf lo = *(const v8bf*)(rowPtr + k0 + kb);
  v8bf hi = *(const v8bf*)(rowPtr + k0 + kb + 16);
  v16bf a;
#pragma unroll
  for (int e = 0; e < 8; ++e) { a[e] = lo[e]; a[8 + e] = hi[e]; }
  return a;
}

// B (32x16, 16-bit) from transposed weights WT[N][K]:
//   lanes 0-15 -> N=lane,     K = k0+{0..7, 16..23}
//   lanes 16-31 -> N=lane-16, K = k0+{8..15, 24..31}
__device__ __forceinline__ v16bf load_bT(const __bf16* WT, int ldk, int n,
                                         int k0, int lane) {
  const int kb = (lane & 16) ? 8 : 0;
  const __bf16* q = WT + (size_t)n * ldk + k0 + kb;
  v8bf lo = *(const v8bf*)q;
  v8bf hi = *(const v8bf*)(q + 16);
  v16bf b;
#pragma unroll
  for (int e = 0; e < 8; ++e) { b[e] = lo[e]; b[8 + e] = hi[e]; }
  return b;
}

// ---- Phase 0: dst[N][K] = (bf16) src[K][N] -------------------------------
__global__ __launch_bounds__(256) void transpose_bf16_kernel(
    const float* __restrict__ src, __bf16* __restrict__ dst, int K, int N) {
  int idx = blockIdx.x * blockDim.x + threadIdx.x;
  int total = K * N;
  if (idx < total) {
    int n = idx / K;
    int k = idx - n * K;
    dst[idx] = (__bf16)src[(size_t)k * N + n];
  }
}

// ---- Phase 1: xw[T][B][H] = X @ W1 + b -----------------------------------
// grid: (M/16), block 256 (8 waves). Wave w: two 16x16 tiles (cols w*32..w*32+31),
// one shared A fragment feeding two independent WMMA chains.
__global__ __launch_bounds__(256) void gemm_xw_kernel(
    const float* __restrict__ X,     // [B][T][D]
    const __bf16* __restrict__ W1T,  // [H][D]
    const float* __restrict__ bias,  // [H]
    float* __restrict__ XW)          // [T*B][H], row r = t*64+b
{
  const int lane = threadIdx.x & 31;
  const int wave = threadIdx.x >> 5;
  const int r0 = blockIdx.x * 16;          // time-first row base (fixed t)
  const int t = r0 >> 6;
  const int b0 = r0 & 63;
  const int m = lane & 15;
  const int n0 = wave * 32 + m;
  const int n1 = n0 + 16;
  const int rbase = (lane & 16) ? 8 : 0;

  const float* arow = X + ((size_t)(b0 + m) * T_DIM + t) * D_DIM;

  v8f acc0, acc1;
  const float bv0 = bias[n0];
  const float bv1 = bias[n1];
#pragma unroll
  for (int i = 0; i < 8; ++i) { acc0[i] = bv0; acc1[i] = bv1; }

#pragma unroll
  for (int k0 = 0; k0 < D_DIM; k0 += 32) {
    v16bf a  = load_a_f32(arow, k0, lane);
    v16bf bf0 = load_bT(W1T, D_DIM, n0, k0, lane);
    v16bf bf1 = load_bT(W1T, D_DIM, n1, k0, lane);
    acc0 = WMMA_BF16(a, bf0, acc0);
    acc1 = WMMA_BF16(a, bf1, acc1);
  }

  float* out0 = XW + (size_t)r0 * H_DIM + n0;
  float* out1 = XW + (size_t)r0 * H_DIM + n1;
#pragma unroll
  for (int i = 0; i < 8; ++i) {
    out0[(size_t)(rbase + i) * H_DIM] = acc0[i];
    out1[(size_t)(rbase + i) * H_DIM] = acc1[i];
  }
}

// ---- Phase 2: sequential scan --------------------------------------------
// grid: 4 blocks, 256 threads (8 waves). Block g owns batch rows g*16..g*16+15.
__global__ __launch_bounds__(256) void rnn_scan_kernel(
    const float* __restrict__ XW,    // [T][B][H]
    const __bf16* __restrict__ W2T,  // [H][H]  (N-major)
    const __bf16* __restrict__ VT,   // [C][H]  (N-major)
    const float* __restrict__ cbias, // [C]
    float* __restrict__ Y,           // [B][T][C]
    float* __restrict__ S)           // [B][T][H]
{
  __shared__ __bf16 w2s[H_DIM * H_DIM];      // 128 KB
  __shared__ __bf16 vs[C_DIM * H_DIM];       //  32 KB
  __shared__ __bf16 hbuf[2][16 * H_DIM];     //  16 KB (double-buffered h)
  __shared__ float  llogp[2][16 * C_DIM];    //   8 KB (K-split partials)

  const int tid = threadIdx.x;
  const int lane = tid & 31;
  const int wave = tid >> 5;
  const int b0 = blockIdx.x * 16;
  const int m = lane & 15;
  const int rbase = (lane & 16) ? 8 : 0;

  // One-time: stage weights into LDS, zero h0.
  {
    const uint32_t* s1 = (const uint32_t*)W2T;
    uint32_t* d1 = (uint32_t*)w2s;
    for (int i = tid; i < H_DIM * H_DIM / 2; i += 256) d1[i] = s1[i];
    const uint32_t* s2 = (const uint32_t*)VT;
    uint32_t* d2 = (uint32_t*)vs;
    for (int i = tid; i < C_DIM * H_DIM / 2; i += 256) d2[i] = s2[i];
    uint32_t* hz = (uint32_t*)&hbuf[0][0];
    for (int i = tid; i < 16 * H_DIM / 2; i += 256) hz[i] = 0u;
  }
  __syncthreads();

  // Step-invariant: this wave's V fragments (logits tile = wave&3, K-half = wave>>2)
  const int cn = (wave & 3) * 16 + m;       // logits column
  const int kh = wave >> 2;                 // K half: 0 or 1
  v16bf vfrag[4];
#pragma unroll
  for (int j = 0; j < 4; ++j)
    vfrag[j] = load_bT(vs, H_DIM, cn, kh * 128 + j * 32, lane);

  // This wave's two h-tile columns
  const int n0 = (wave * 2 + 0) * 16 + m;
  const int n1 = (wave * 2 + 1) * 16 + m;

  // Software-pipelined xw accumulator-init registers
  float xcur[2][8], xnxt[2][8];
  {
    const float* xw0 = XW + (size_t)b0 * H_DIM;   // t = 0
#pragma unroll
    for (int i = 0; i < 8; ++i) {
      xcur[0][i] = xw0[(size_t)(rbase + i) * H_DIM + n0];
      xcur[1][i] = xw0[(size_t)(rbase + i) * H_DIM + n1];
    }
  }

  int p = 0;
  for (int t = 0; t < T_DIM; ++t) {
    // Issue next-step xw loads now; consumed only at end of this iteration.
    if (t + 1 < T_DIM) {
      const float* xwn = XW + ((size_t)(t + 1) * B_DIM + b0) * H_DIM;
#pragma unroll
      for (int i = 0; i < 8; ++i) {
        xnxt[0][i] = xwn[(size_t)(rbase + i) * H_DIM + n0];
        xnxt[1][i] = xwn[(size_t)(rbase + i) * H_DIM + n1];
      }
    }
    // Speculative L2 prefetch of the t+2 slab (16 KB / 256 threads = 64 B each)
    if (t + 2 < T_DIM) {
      const char* pf = (const char*)(XW + ((size_t)(t + 2) * B_DIM + b0) * H_DIM) + tid * 64;
      __builtin_prefetch(pf, 0, 1);
    }

    // --- h_new = tanh(xw_t + h @ W2): one shared A fragment, two WMMA chains ---
    {
      v8f acc0, acc1;
#pragma unroll
      for (int i = 0; i < 8; ++i) { acc0[i] = xcur[0][i]; acc1[i] = xcur[1][i]; }

      const __bf16* arow = &hbuf[p][m * H_DIM];
#pragma unroll
      for (int k0 = 0; k0 < H_DIM; k0 += 32) {
        v16bf a   = load_a_bf16(arow, k0, lane);
        v16bf bf0 = load_bT(w2s, H_DIM, n0, k0, lane);
        v16bf bf1 = load_bT(w2s, H_DIM, n1, k0, lane);
        acc0 = WMMA_BF16(a, bf0, acc0);
        acc1 = WMMA_BF16(a, bf1, acc1);
      }
#pragma unroll
      for (int i = 0; i < 8; ++i) {
        const int brow = rbase + i;
        const float hv0 = fast_tanhf(acc0[i]);
        const float hv1 = fast_tanhf(acc1[i]);
        float* srow = S + (((size_t)(b0 + brow)) * T_DIM + t) * H_DIM;
        srow[n0] = hv0;
        srow[n1] = hv1;
        hbuf[1 - p][brow * H_DIM + n0] = (__bf16)hv0;
        hbuf[1 - p][brow * H_DIM + n1] = (__bf16)hv1;
      }
    }
    __syncthreads();   // (1) h_new visible in hbuf[1-p]; llogp free to overwrite

    // --- logits = c + h_new @ V; all 8 waves, K split in halves ---
    {
      v8f acc;
      const float cb = (kh == 0) ? cbias[cn] : 0.0f;
#pragma unroll
      for (int i = 0; i < 8; ++i) acc[i] = cb;

      const __bf16* arow = &hbuf[1 - p][m * H_DIM];
#pragma unroll
      for (int j = 0; j < 4; ++j) {
        v16bf a = load_a_bf16(arow, kh * 128 + j * 32, lane);
        acc = WMMA_BF16(a, vfrag[j], acc);
      }
#pragma unroll
      for (int i = 0; i < 8; ++i) llogp[kh][(rbase + i) * C_DIM + cn] = acc[i];
    }
    __syncthreads();   // (2) logits partials visible

    // --- rowwise softmax over llogp sums; 16 threads per row; no trailing
    //     barrier: next step's llogp writes sit behind next barrier (1). ---
    {
      const int row = tid >> 4;   // 0..15 (16 contiguous lanes per row)
      const int l16 = tid & 15;
      const int base = row * C_DIM + l16 * 4;
      float v0 = llogp[0][base + 0] + llogp[1][base + 0];
      float v1 = llogp[0][base + 1] + llogp[1][base + 1];
      float v2 = llogp[0][base + 2] + llogp[1][base + 2];
      float v3 = llogp[0][base + 3] + llogp[1][base + 3];
      float mx = fmaxf(fmaxf(v0, v1), fmaxf(v2, v3));
#pragma unroll
      for (int o = 8; o >= 1; o >>= 1) mx = fmaxf(mx, __shfl_xor(mx, o, 32));
      float e0 = __expf(v0 - mx), e1 = __expf(v1 - mx);
      float e2 = __expf(v2 - mx), e3 = __expf(v3 - mx);
      float sm = e0 + e1 + e2 + e3;
#pragma unroll
      for (int o = 8; o >= 1; o >>= 1) sm += __shfl_xor(sm, o, 32);
      const float inv = 1.0f / sm;
      float* yp = Y + (((size_t)(b0 + row)) * T_DIM + t) * C_DIM + l16 * 4;
      yp[0] = e0 * inv; yp[1] = e1 * inv; yp[2] = e2 * inv; yp[3] = e3 * inv;
    }

    // Rotate software pipeline
#pragma unroll
    for (int i = 0; i < 8; ++i) {
      xcur[0][i] = xnxt[0][i];
      xcur[1][i] = xnxt[1][i];
    }
    p ^= 1;
  }
}

// ---------------------------------------------------------------------------
extern "C" void kernel_launch(void* const* d_in, const int* in_sizes, int n_in,
                              void* d_out, int out_size, void* d_ws, size_t ws_size,
                              hipStream_t stream) {
  const float* X  = (const float*)d_in[0];  // [B,T,D]
  const float* W1 = (const float*)d_in[1];  // [D,H]
  const float* W2 = (const float*)d_in[2];  // [H,H]
  const float* bb = (const float*)d_in[3];  // [H]
  const float* V  = (const float*)d_in[4];  // [H,C]
  const float* cc = (const float*)d_in[5];  // [C]

  float* Y = (float*)d_out;                                   // [B,T,C]
  float* S = Y + (size_t)B_DIM * T_DIM * C_DIM;               // [B,T,H]

  // Workspace: xw (128 MB f32) + bf16 weight copies (~288 KB)
  char* ws = (char*)d_ws;
  float* XW = (float*)ws;
  size_t off = (size_t)T_DIM * B_DIM * H_DIM * sizeof(float);
  __bf16* W1T = (__bf16*)(ws + off); off += (size_t)H_DIM * D_DIM * sizeof(__bf16);
  __bf16* W2T = (__bf16*)(ws + off); off += (size_t)H_DIM * H_DIM * sizeof(__bf16);
  __bf16* VT  = (__bf16*)(ws + off); off += (size_t)C_DIM * H_DIM * sizeof(__bf16);

  // Phase 0: weight transpose + bf16 convert
  transpose_bf16_kernel<<<(D_DIM * H_DIM + 255) / 256, 256, 0, stream>>>(W1, W1T, D_DIM, H_DIM);
  transpose_bf16_kernel<<<(H_DIM * H_DIM + 255) / 256, 256, 0, stream>>>(W2, W2T, H_DIM, H_DIM);
  transpose_bf16_kernel<<<(H_DIM * C_DIM + 255) / 256, 256, 0, stream>>>(V,  VT,  H_DIM, C_DIM);

  // Phase 1: input projection GEMM (M = T*B rows, time-first)
  gemm_xw_kernel<<<(T_DIM * B_DIM) / 16, 256, 0, stream>>>(X, W1T, bb, XW);

  // Phase 2: sequential scan, batch-parallel across 4 workgroups
  rnn_scan_kernel<<<4, 256, 0, stream>>>(XW, W2T, VT, cc, Y, S);
}
